// graph_constructor_53128745451587
// MI455X (gfx1250) — matrix operand order repaired and checked by reading, not compiled, not run on hardware
//
#include <hip/hip_runtime.h>

#define NND 8192
#define DD 128
#define KSEL 32
#define ALPHA_F 3.0f
#define LDS_STRIDE 136  // 128 halves + 8 halves (16B) pad -> 272B row stride, bank-conflict-free

typedef __attribute__((ext_vector_type(8)))  _Float16 v8h;
typedef __attribute__((ext_vector_type(16))) _Float16 v16h;
typedef __attribute__((ext_vector_type(8)))  float    v8f;
typedef __attribute__((ext_vector_type(4)))  unsigned int u32x4;
typedef __attribute__((ext_vector_type(8)))  int          i32x8;
typedef __attribute__((ext_vector_type(4)))  int          i32x4;

// ---------------------------------------------------------------------------
// WMMA fragment loaders for D = X @ Y^T  (operands row-major [rows][128])
// A 16x32 f16: lane L(0..15) row M=L, K=0..15 contiguous;
//   lane L+16: K=16..31.
// B 32x16 f16: lane L(0..15) col N=L, K=0..15 contiguous; lane L+16: K=16..31.
// ---------------------------------------------------------------------------

__device__ __forceinline__ v16h load_fragA_f16(const _Float16* mat, int rowBase,
                                               int kbase, int lane) {
  int r = rowBase + (lane & 15);
  int kofs = kbase + ((lane >> 4) << 4);  // +0 or +16
  const _Float16* p = mat + r * DD + kofs;
  v8h lo = *(const v8h*)p;         // K = kofs .. kofs+7
  v8h hi = *(const v8h*)(p + 8);   // K = kofs+8 .. kofs+15
  v16h out;
#pragma unroll
  for (int i = 0; i < 8; ++i) { out[i] = lo[i]; out[i + 8] = hi[i]; }
  return out;
}

// B-fragment from the LDS-staged (padded) column slab
__device__ __forceinline__ v16h load_fragB_lds(const _Float16* sm, int colLocalBase,
                                               int kbase, int lane) {
  int c = colLocalBase + (lane & 15);
  int kofs = kbase + ((lane >> 4) << 4);  // +0 or +16
  const _Float16* p = sm + c * LDS_STRIDE + kofs;
  v8h lo = *(const v8h*)p;       // 16B aligned (272B row stride, kofs*2 % 16 == 0)
  v8h hi = *(const v8h*)(p + 8);
  v16h out;
#pragma unroll
  for (int i = 0; i < 8; ++i) { out[i] = lo[i]; out[i + 8] = hi[i]; }
  return out;
}

// f32-source variants (convert on load) for the small MLP matmul
__device__ __forceinline__ v16h load_fragA_f32_idx(const float* mat, const int* idx,
                                                   int rowBase, int kbase, int lane) {
  int r = idx[rowBase + (lane & 15)];
  int kofs = kbase + ((lane >> 4) << 4);
  const float* p = mat + r * DD + kofs;
  v16h out;
#pragma unroll
  for (int i = 0; i < 8; ++i) {
    out[i] = (_Float16)p[i];
    out[i + 8] = (_Float16)p[8 + i];
  }
  return out;
}

__device__ __forceinline__ v16h load_fragB_f32(const float* mat, int colBase,
                                               int kbase, int lane) {
  int c = colBase + (lane & 15);
  int kofs = kbase + ((lane >> 4) << 4);
  const float* p = mat + c * DD + kofs;
  v16h out;
#pragma unroll
  for (int i = 0; i < 16; ++i) out[i] = (_Float16)p[i];
  return out;
}

__device__ __forceinline__ float tanh_fast(float x) {
  // tanh(x) = 1 - 2/(exp(2x)+1); stable at both extremes
  return 1.0f - 2.0f / (__expf(2.0f * x) + 1.0f);
}

// deterministic hash noise in [0, 0.01) standing in for jax uniform tie-break
__device__ __forceinline__ float noise01(unsigned row, unsigned col) {
  unsigned h = row * 2654435761u ^ (col * 0x9e3779b9u);
  h ^= h >> 16; h *= 0x7feb352du;
  h ^= h >> 15; h *= 0x846ca68bu;
  h ^= h >> 16;
  return (float)(h >> 8) * (0.01f / 16777216.0f);
}

// ---------------------------------------------------------------------------
// TDM: DMA a 64-row x 128-half f16 slab (row-major, stride 128) into LDS,
// inserting 16B of LDS padding after every 256B row (-> 272B LDS row stride).
// D# built per CDNA5 ISA; 2D tile so groups 2/3
// are zero. Tracked by TENSORcnt. 6-arg builtin form (clang-23 toolchain).
// ---------------------------------------------------------------------------
__device__ __forceinline__ void tdm_load_slab64x128_f16(const _Float16* gsrc,
                                                        unsigned lds_byte_addr) {
  unsigned long long ga = (unsigned long long)(uintptr_t)gsrc;
  u32x4 g0;
  g0[0] = 1u;                                      // count=1, user D#, no gather
  g0[1] = lds_byte_addr;                           // lds_addr (bits 63:32)
  g0[2] = (unsigned)(ga & 0xFFFFFFFFull);          // global_addr[31:0]
  g0[3] = (unsigned)((ga >> 32) & 0x01FFFFFFull)   // global_addr[56:32]
        | (2u << 30);                              // type=2 ("image")

  i32x8 g1;
  // word0: wg_mask=0 | data_size=1(2B)@16 | pad_enable@20 | pad_interval=5(64dw)@22
  //        | pad_amount=3(4dw)@25
  g1[0] = (1 << 16) | (1 << 20) | (5 << 22) | (3 << 25);
  g1[1] = (int)(128u << 16);    // tensor_dim0 = 128 (low 16 bits @ word1[31:16])
  g1[2] = (int)(8192u << 16);   // tensor_dim0 hi=0; tensor_dim1=8192 low16 @ word2[31:16]
  g1[3] = (int)(128u << 16);    // tensor_dim1 hi=0; tile_dim0=128 @ word3[31:16]
  g1[4] = 64;                   // tile_dim1 = 64; tile_dim2 = 0
  g1[5] = 128;                  // tensor_dim0_stride = 128 (low 32)
  g1[6] = 0;                    // dim0_stride hi; dim1_stride low (unused for 2D)
  g1[7] = 0;

  i32x4 gz4 = {0, 0, 0, 0};
  i32x8 gz8 = {0, 0, 0, 0, 0, 0, 0, 0};
  __builtin_amdgcn_tensor_load_to_lds(g0, g1, gz4, gz4, gz8, 0);
}

// ---------------------------------------------------------------------------
// Kernel 1: n = tanh(ALPHA * (emb[idx] @ w^T + b)) stored as f16 [NND][DD]
// ---------------------------------------------------------------------------
__global__ void __launch_bounds__(32)
embed_linear_tanh_kernel(const int* __restrict__ idx, const float* __restrict__ emb,
                         const float* __restrict__ w, const float* __restrict__ b,
                         _Float16* __restrict__ nout) {
  int lane = threadIdx.x & 31;
  int rowBase = blockIdx.x * 16;

  v8f acc[8];
#pragma unroll
  for (int j = 0; j < 8; ++j)
#pragma unroll
    for (int v = 0; v < 8; ++v) acc[j][v] = 0.0f;

#pragma unroll
  for (int k = 0; k < DD; k += 32) {
    v16h a = load_fragA_f32_idx(emb, idx, rowBase, k, lane);
#pragma unroll
    for (int j = 0; j < 8; ++j) {
      v16h bb = load_fragB_f32(w, j * 16, k, lane);
      acc[j] = __builtin_amdgcn_wmma_f32_16x16x32_f16(false, a, false, bb,
                                                      (short)0, acc[j], false, false);
    }
  }

  int m0 = rowBase + ((lane >> 4) << 3);
  int n0 = lane & 15;
#pragma unroll
  for (int j = 0; j < 8; ++j) {
    float bias = b[j * 16 + n0];
#pragma unroll
    for (int v = 0; v < 8; ++v) {
      float t = tanh_fast(ALPHA_F * (acc[j][v] + bias));
      nout[(m0 + v) * DD + j * 16 + n0] = (_Float16)t;
    }
  }
}

// ---------------------------------------------------------------------------
// Kernel 2: adj = relu(tanh(ALPHA * (n1@n2^T - n2@n1^T)))   [NND][NND] f32
// 128x64 tile / 8-wave block. Column slabs of n1/n2 staged into LDS by TDM
// (wave 0), A-fragments stream from L2-resident global (4MB working set).
// ---------------------------------------------------------------------------
__global__ void __launch_bounds__(256)
antisym_gemm_kernel(const _Float16* __restrict__ n1, const _Float16* __restrict__ n2,
                    float* __restrict__ out) {
  __shared__ _Float16 smB[2][64 * LDS_STRIDE];  // [0]=n2 slab, [1]=n1 slab (~34KB)

  int lane = threadIdx.x & 31;
  int wave = threadIdx.x >> 5;
  int rowBase = blockIdx.y * 128 + wave * 16;
  int colBase = blockIdx.x * 64;

  if (wave == 0) {
    unsigned lds0 = (unsigned)(uintptr_t)(&smB[0][0]);
    unsigned lds1 = (unsigned)(uintptr_t)(&smB[1][0]);
    tdm_load_slab64x128_f16(n2 + (size_t)colBase * DD, lds0);
    tdm_load_slab64x128_f16(n1 + (size_t)colBase * DD, lds1);
    __builtin_amdgcn_s_wait_tensorcnt(0);
  }
  __syncthreads();

  v8f acc1[4], acc2[4];
#pragma unroll
  for (int j = 0; j < 4; ++j)
#pragma unroll
    for (int v = 0; v < 8; ++v) { acc1[j][v] = 0.0f; acc2[j][v] = 0.0f; }

#pragma unroll
  for (int k = 0; k < DD; k += 32) {
    v16h a1 = load_fragA_f16(n1, rowBase, k, lane);
    v16h a2 = load_fragA_f16(n2, rowBase, k, lane);
#pragma unroll
    for (int j = 0; j < 4; ++j) {
      v16h b1 = load_fragB_lds(&smB[0][0], j * 16, k, lane);  // n2 cols
      v16h b2 = load_fragB_lds(&smB[1][0], j * 16, k, lane);  // n1 cols
      acc1[j] = __builtin_amdgcn_wmma_f32_16x16x32_f16(false, a1, false, b1,
                                                       (short)0, acc1[j], false, false);
      acc2[j] = __builtin_amdgcn_wmma_f32_16x16x32_f16(false, a2, false, b2,
                                                       (short)0, acc2[j], false, false);
    }
  }

  int m0 = rowBase + ((lane >> 4) << 3);
  int n0 = lane & 15;
#pragma unroll
  for (int j = 0; j < 4; ++j) {
#pragma unroll
    for (int v = 0; v < 8; ++v) {
      float aval = acc1[j][v] - acc2[j][v];
      float adj = 0.0f;
      if (aval > 0.0f) {
        float e = __expf(-2.0f * ALPHA_F * aval);
        adj = (1.0f - e) / (1.0f + e);  // tanh(ALPHA*aval), aval>0
      }
      out[(size_t)(m0 + v) * NND + colBase + j * 16 + n0] = adj;
    }
  }
}

// ---------------------------------------------------------------------------
// Kernel 3: in-place per-row top-K(32) mask on (adj + noise), exact radix
// select (MSB-first, 4x8-bit), deterministic lowest-index tie-break.
// ---------------------------------------------------------------------------
__global__ void __launch_bounds__(256)
topk_mask_kernel(float* __restrict__ out) {
  int row = blockIdx.x;
  int tid = threadIdx.x;
  float* rowp = out + (size_t)row * NND;
  int base = tid * 32;

  float a[32];
  unsigned u[32];
#pragma unroll
  for (int i = 0; i < 8; ++i) {
    float4 v = *(const float4*)(rowp + base + i * 4);
    a[i * 4 + 0] = v.x; a[i * 4 + 1] = v.y;
    a[i * 4 + 2] = v.z; a[i * 4 + 3] = v.w;
  }
#pragma unroll
  for (int i = 0; i < 32; ++i)
    u[i] = __float_as_uint(a[i] + noise01((unsigned)row, (unsigned)(base + i)));
  // all keys >= 0 -> IEEE bits order-preserving as unsigned

  __shared__ unsigned hist[256];
  __shared__ unsigned texcl[256];
  __shared__ int s_digit, s_rem;

  unsigned prefix = 0, pmask = 0;
  int remK = KSEL;
#pragma unroll 1
  for (int pass = 3; pass >= 0; --pass) {
    int shift = pass * 8;
    hist[tid] = 0;
    __syncthreads();
#pragma unroll
    for (int i = 0; i < 32; ++i)
      if ((u[i] & pmask) == prefix)
        atomicAdd(&hist[(u[i] >> shift) & 255], 1u);
    __syncthreads();
    if (tid == 0) {
      unsigned c = 0;
      int d = 255;
      for (; d > 0; --d) {
        if (c + hist[d] >= (unsigned)remK) break;
        c += hist[d];
      }
      s_digit = d;
      s_rem = remK - (int)c;
    }
    __syncthreads();
    prefix |= ((unsigned)s_digit) << shift;
    pmask |= 255u << shift;
    remK = s_rem;
    __syncthreads();
  }
  unsigned T = prefix;  // exact K-th largest key; remK = #ties of T to keep

  unsigned mine = 0;
#pragma unroll
  for (int i = 0; i < 32; ++i) mine += (u[i] == T) ? 1u : 0u;
  hist[tid] = mine;
  __syncthreads();
  if (tid == 0) {
    unsigned run = 0;
    for (int t = 0; t < 256; ++t) { texcl[t] = run; run += hist[t]; }
  }
  __syncthreads();

  int trank = (int)texcl[tid];
#pragma unroll
  for (int i = 0; i < 32; ++i) {
    bool keep = false;
    if (u[i] > T) keep = true;
    else if (u[i] == T) { keep = (trank < remK); ++trank; }
    if (!keep) a[i] = 0.0f;
  }
#pragma unroll
  for (int i = 0; i < 8; ++i) {
    float4 v;
    v.x = a[i * 4 + 0]; v.y = a[i * 4 + 1];
    v.z = a[i * 4 + 2]; v.w = a[i * 4 + 3];
    *(float4*)(rowp + base + i * 4) = v;
  }
}

// ---------------------------------------------------------------------------
extern "C" void kernel_launch(void* const* d_in, const int* in_sizes, int n_in,
                              void* d_out, int out_size, void* d_ws, size_t ws_size,
                              hipStream_t stream) {
  const int*   idx  = (const int*)d_in[0];
  const float* emb1 = (const float*)d_in[1];
  const float* emb2 = (const float*)d_in[2];
  const float* w1   = (const float*)d_in[3];
  const float* b1   = (const float*)d_in[4];
  const float* w2   = (const float*)d_in[5];
  const float* b2   = (const float*)d_in[6];
  float* out = (float*)d_out;

  _Float16* n1 = (_Float16*)d_ws;                 // [NND][DD] f16, 2MB
  _Float16* n2 = n1 + (size_t)NND * DD;           // [NND][DD] f16, 2MB

  embed_linear_tanh_kernel<<<dim3(NND / 16), 32, 0, stream>>>(idx, emb1, w1, b1, n1);
  embed_linear_tanh_kernel<<<dim3(NND / 16), 32, 0, stream>>>(idx, emb2, w2, b2, n2);

  antisym_gemm_kernel<<<dim3(NND / 64, NND / 128), 256, 0, stream>>>(n1, n2, out);

  topk_mask_kernel<<<dim3(NND), 256, 0, stream>>>(out);
}
